// MultiHeadAttention_46420006535638
// MI455X (gfx1250) — matrix operand order repaired and checked by reading, not compiled, not run on hardware
//
#include <hip/hip_runtime.h>
#include <hip/hip_bf16.h>

#define EMB   1024
#define NHEAD 16
#define HDIM  64
#define BATCH 2
#define SEQ   2048
#define BS    (BATCH * SEQ)

typedef __bf16 bf16_t;
typedef __attribute__((ext_vector_type(16))) __bf16 v16bf;
typedef __attribute__((ext_vector_type(8)))  __bf16 v8bf;
typedef __attribute__((ext_vector_type(8)))  float  v8f;

__device__ __forceinline__ v16bf cat8(v8bf lo, v8bf hi) {
  return __builtin_shufflevector(lo, hi, 0,1,2,3,4,5,6,7,8,9,10,11,12,13,14,15);
}
__device__ __forceinline__ v8f wmma_bf16(v16bf a, v16bf b, v8f c) {
  return __builtin_amdgcn_wmma_f32_16x16x32_bf16(false, a, false, b, (short)0, c, false, false);
}

// ---------------------------------------------------------------- fp32 -> bf16
__global__ void k_cvt_x(const float* __restrict__ x, bf16_t* __restrict__ xb) {
  int i = (blockIdx.x * blockDim.x + threadIdx.x) * 4;
  float4 v = *(const float4*)(x + i);
  xb[i + 0] = (bf16_t)v.x;
  xb[i + 1] = (bf16_t)v.y;
  xb[i + 2] = (bf16_t)v.z;
  xb[i + 3] = (bf16_t)v.w;
}

// ------------------------------------------- weights: [in][out] -> bf16 [out][in]
__global__ void k_transpose_w(const float* __restrict__ Wq, const float* __restrict__ Wk,
                              const float* __restrict__ Wv, const float* __restrict__ Wo,
                              bf16_t* __restrict__ Tq, bf16_t* __restrict__ Tk,
                              bf16_t* __restrict__ Tv, bf16_t* __restrict__ To) {
  __shared__ float tile[16][17];
  const float* W; bf16_t* T;
  switch (blockIdx.z) {
    case 0:  W = Wq; T = Tq; break;
    case 1:  W = Wk; T = Tk; break;
    case 2:  W = Wv; T = Tv; break;
    default: W = Wo; T = To; break;
  }
  int k0 = blockIdx.x * 16, n0 = blockIdx.y * 16;
  tile[threadIdx.y][threadIdx.x] = W[(size_t)(k0 + threadIdx.y) * EMB + n0 + threadIdx.x];
  __syncthreads();
  T[(size_t)(n0 + threadIdx.y) * EMB + k0 + threadIdx.x] = (bf16_t)tile[threadIdx.x][threadIdx.y];
}

// ------------------------------------------------------------------ WMMA GEMM
// C[32x64] per wave: every B (weight) fragment is reused by two A fragments,
// so each 32-wide k-step issues 8 WMMAs for 12 x b128 loads.
// mode 0: Q -> [B,H,S,D] bf16   mode 1: K -> [B,H,S,D] bf16
// mode 2: V -> [B,H,D,S] bf16   mode 3: out -> [BS,EMB] f32
__global__ void k_gemm(const bf16_t* __restrict__ A, const bf16_t* __restrict__ Wt,
                       const float* __restrict__ bias, void* __restrict__ dst, int mode) {
  int lane = threadIdx.x;
  int hf = lane >> 4, ln = lane & 15;
  int m0 = blockIdx.x * 32, n0 = blockIdx.y * 64;

  const bf16_t* arow0 = A + (size_t)(m0 + ln) * EMB;
  const bf16_t* arow1 = A + (size_t)(m0 + 16 + ln) * EMB;
  v8f acc[2][4] = {};

  for (int k0 = 0; k0 < EMB; k0 += 32) {
    // prefetch the next k-block of the streamed A rows (global_prefetch_b8)
    if (k0 + 64 <= EMB) {
      __builtin_prefetch(arow0 + k0 + 32, 0, 1);
      __builtin_prefetch(arow1 + k0 + 32, 0, 1);
    }
    v16bf af0 = cat8(*(const v8bf*)(arow0 + k0 + hf * 8),
                     *(const v8bf*)(arow0 + k0 + 16 + hf * 8));
    v16bf af1 = cat8(*(const v8bf*)(arow1 + k0 + hf * 8),
                     *(const v8bf*)(arow1 + k0 + 16 + hf * 8));
#pragma unroll
    for (int t = 0; t < 4; ++t) {
      const bf16_t* bp = Wt + (size_t)(n0 + t * 16 + ln) * EMB + k0;
      v16bf bfrag = cat8(*(const v8bf*)(bp + hf * 8), *(const v8bf*)(bp + 16 + hf * 8));
      acc[0][t] = wmma_bf16(af0, bfrag, acc[0][t]);
      acc[1][t] = wmma_bf16(af1, bfrag, acc[1][t]);
    }
  }

#pragma unroll
  for (int mi = 0; mi < 2; ++mi) {
#pragma unroll
    for (int t = 0; t < 4; ++t) {
      int n = n0 + t * 16 + ln;
      float bv = bias[n];
      int hh = n / HDIM, d = n % HDIM;
#pragma unroll
      for (int r = 0; r < 8; ++r) {
        int sm = m0 + mi * 16 + r + 8 * hf;         // global row in [0, BS)
        float val = acc[mi][t][r] + bv;
        if (mode == 3) {
          ((float*)dst)[(size_t)sm * EMB + n] = val;
        } else {
          int bb = sm / SEQ, s = sm % SEQ;
          bf16_t* o = (bf16_t*)dst;
          if (mode == 2)
            o[(((size_t)bb * NHEAD + hh) * HDIM + d) * SEQ + s] = (bf16_t)val;  // V^T
          else
            o[(((size_t)bb * NHEAD + hh) * SEQ + s) * HDIM + d] = (bf16_t)val;  // Q / K
        }
      }
    }
  }
}

// --------------------------------------------------------- causal flash attention
// One wave per (b, head, 16 query rows); 32 keys per iteration.
__global__ void k_attn(const bf16_t* __restrict__ Q, const bf16_t* __restrict__ K,
                       const bf16_t* __restrict__ Vt, bf16_t* __restrict__ ctx) {
  __shared__ __align__(32) bf16_t p_lds[16 * 32];
  int lane = threadIdx.x;
  int hf = lane >> 4, ln = lane & 15;
  int qt = blockIdx.x, head = blockIdx.y, bb = blockIdx.z;
  int bh = bb * NHEAD + head;
  int qbase = qt * 16;

  // Q A-fragments for d in [0,32) and [32,64), reused across the whole key loop
  const bf16_t* qrow = Q + ((size_t)bh * SEQ + qbase + ln) * HDIM;
  v16bf a0 = cat8(*(const v8bf*)(qrow + hf * 8),      *(const v8bf*)(qrow + 16 + hf * 8));
  v16bf a1 = cat8(*(const v8bf*)(qrow + 32 + hf * 8), *(const v8bf*)(qrow + 48 + hf * 8));

  float m[8], l[8], corr[8];
#pragma unroll
  for (int r = 0; r < 8; ++r) { m[r] = -1e30f; l[r] = 0.f; }
  v8f acc[4] = {};

  const float scale = 0.125f;                    // 1/sqrt(HDIM)
  for (int kb = 0; kb < qbase + 16; kb += 32) {
    // ---- scores: two 16-key sub-tiles, K-dim 64 = two WMMA each
    const bf16_t* k0p = K + ((size_t)bh * SEQ + kb + ln) * HDIM;
    const bf16_t* k1p = k0p + (size_t)16 * HDIM;
    // prefetch the next 32-key block of K (global_prefetch_b8)
    if (kb + 32 < qbase + 16) __builtin_prefetch(k0p + (size_t)32 * HDIM, 0, 1);
    v8f s0 = {}, s1 = {};
    s0 = wmma_bf16(a0, cat8(*(const v8bf*)(k0p + hf * 8),      *(const v8bf*)(k0p + 16 + hf * 8)), s0);
    s0 = wmma_bf16(a1, cat8(*(const v8bf*)(k0p + 32 + hf * 8), *(const v8bf*)(k0p + 48 + hf * 8)), s0);
    s1 = wmma_bf16(a0, cat8(*(const v8bf*)(k1p + hf * 8),      *(const v8bf*)(k1p + 16 + hf * 8)), s1);
    s1 = wmma_bf16(a1, cat8(*(const v8bf*)(k1p + 32 + hf * 8), *(const v8bf*)(k1p + 48 + hf * 8)), s1);

    // ---- online softmax over this 16x32 block (row = r + 8*hf, col = lane&15)
#pragma unroll
    for (int r = 0; r < 8; ++r) {
      int qi = qbase + r + 8 * hf;
      float v0 = (kb + ln      <= qi) ? s0[r] * scale : -1e30f;
      float v1 = (kb + 16 + ln <= qi) ? s1[r] * scale : -1e30f;
      float rmax = fmaxf(v0, v1);
#pragma unroll
      for (int off = 1; off < 16; off <<= 1)
        rmax = fmaxf(rmax, __shfl_xor(rmax, off));
      float mnew = fmaxf(m[r], rmax);
      float p0 = __expf(v0 - mnew);
      float p1 = __expf(v1 - mnew);
      float rsum = p0 + p1;
#pragma unroll
      for (int off = 1; off < 16; off <<= 1)
        rsum += __shfl_xor(rsum, off);
      float c = __expf(m[r] - mnew);
      l[r] = l[r] * c + rsum;
      m[r] = mnew;
      corr[r] = c;
      // stage P (C-layout) into LDS row-major so it can be re-read in A-layout
      p_lds[(r + 8 * hf) * 32 + ln]      = (bf16_t)p0;
      p_lds[(r + 8 * hf) * 32 + 16 + ln] = (bf16_t)p1;
    }
    __syncthreads();

    const bf16_t* prow = p_lds + ln * 32;
    v16bf pf = cat8(*(const v8bf*)(prow + hf * 8), *(const v8bf*)(prow + 16 + hf * 8));

    // ---- rescale running accumulators, then P(16x32) @ V(32x64)
#pragma unroll
    for (int t = 0; t < 4; ++t) {
#pragma unroll
      for (int r = 0; r < 8; ++r) acc[t][r] *= corr[r];
      const bf16_t* vp = Vt + ((size_t)bh * HDIM + t * 16 + ln) * SEQ + kb;
      acc[t] = wmma_bf16(pf, cat8(*(const v8bf*)(vp + hf * 8), *(const v8bf*)(vp + 16 + hf * 8)), acc[t]);
    }
    __syncthreads();
  }

  // ---- normalize and write ctx as bf16 [B, S, E] for the output projection
#pragma unroll
  for (int t = 0; t < 4; ++t)
#pragma unroll
    for (int r = 0; r < 8; ++r) {
      int q = qbase + r + 8 * hf;
      float val = acc[t][r] / l[r];
      ctx[((size_t)bb * SEQ + q) * EMB + head * HDIM + t * 16 + ln] = (bf16_t)val;
    }
}

// --------------------------------------------------------------------- launch
extern "C" void kernel_launch(void* const* d_in, const int* in_sizes, int n_in,
                              void* d_out, int out_size, void* d_ws, size_t ws_size,
                              hipStream_t stream) {
  const float* x  = (const float*)d_in[0];
  const float* Wq = (const float*)d_in[1];
  const float* bq = (const float*)d_in[2];
  const float* Wk = (const float*)d_in[3];
  const float* bk = (const float*)d_in[4];
  const float* Wv = (const float*)d_in[5];
  const float* bv = (const float*)d_in[6];
  const float* Wo = (const float*)d_in[7];
  const float* bo = (const float*)d_in[8];

  char* ws = (char*)d_ws;
  bf16_t* xbf = (bf16_t*)(ws);                    //  8 MiB: x as bf16 [BS, E]
  bf16_t* wtq = (bf16_t*)(ws + ( 8ull << 20));    //  2 MiB each: W^T bf16 [out][in]
  bf16_t* wtk = (bf16_t*)(ws + (10ull << 20));
  bf16_t* wtv = (bf16_t*)(ws + (12ull << 20));
  bf16_t* wto = (bf16_t*)(ws + (14ull << 20));
  bf16_t* qb  = (bf16_t*)(ws + (16ull << 20));    //  8 MiB: Q  [B,H,S,D]
  bf16_t* kb  = (bf16_t*)(ws + (24ull << 20));    //  8 MiB: K  [B,H,S,D]
  bf16_t* vt  = (bf16_t*)(ws + (32ull << 20));    //  8 MiB: V^T [B,H,D,S]
  bf16_t* ctx = (bf16_t*)(ws + (40ull << 20));    //  8 MiB: ctx [B,S,E]

  k_cvt_x<<<(BS * EMB) / (256 * 4), 256, 0, stream>>>(x, xbf);
  k_transpose_w<<<dim3(EMB / 16, EMB / 16, 4), dim3(16, 16, 1), 0, stream>>>(
      Wq, Wk, Wv, Wo, wtq, wtk, wtv, wto);

  dim3 gg(BS / 32, EMB / 64, 1);
  k_gemm<<<gg, 32, 0, stream>>>(xbf, wtq, bq, qb, 0);
  k_gemm<<<gg, 32, 0, stream>>>(xbf, wtk, bk, kb, 1);
  k_gemm<<<gg, 32, 0, stream>>>(xbf, wtv, bv, vt, 2);

  k_attn<<<dim3(SEQ / 16, NHEAD, BATCH), 32, 0, stream>>>(qb, kb, vt, ctx);

  k_gemm<<<gg, 32, 0, stream>>>(ctx, wto, bo, d_out, 3);
}
